// TttBaseModule_41223096107420
// MI455X (gfx1250) — matrix operand order repaired and checked by reading, not compiled, not run on hardware
//
#include <hip/hip_runtime.h>
#include <hip/hip_bf16.h>

// ---------------------------------------------------------------------------
// TTT layer for MI455X (gfx1250, wave32, WMMA).
//   B=8, L=2048, D=2048, NH=32, HD=64, CS=16
// Pipeline:
//   1. convert hidden_states + 4 weight matrices f32 -> bf16
//   2. WMMA bf16 GEMMs (software-pipelined, register-staged tiles):
//      XQ/XK/XV = hs @ w{q,k,v}, output permuted (B,NH,L,HD) f32
//   3. RoPE (interleaved-pair form; equals permute->rope->unpermute) in place
//   4. scan kernel: one workgroup (128 thr = 4 waves) per (b,h); W1/b1 state in
//      LDS; per 16-token chunk all matmuls are v_wmma_f32_16x16x32_bf16 with
//      zero-padded K where the contraction length is 16; LN reductions via
//      __shfl_xor in 8-lane row groups; attention output stored bf16 (B,L,D)
//   5. WMMA bf16 GEMM: out = att @ wo -> d_out (f32)
// ---------------------------------------------------------------------------

typedef __bf16 bf16_t;
typedef __attribute__((ext_vector_type(8)))  __bf16 v8bf;
typedef __attribute__((ext_vector_type(16))) __bf16 v16bf;
typedef __attribute__((ext_vector_type(8)))  float  v8f;

#define TB  8
#define TL  2048
#define TD  2048
#define TNH 32
#define THD 64
#define TCS 16

__device__ __forceinline__ bf16_t f2bf(float f) { return (bf16_t)f; }

// A-matrix fragment (16xK bf16, K>=32 window), row-major source [m][k], ld elems.
// ISA layout: lanes 0-15 -> K kb..kb+7 (v0..3) and kb+16..kb+23 (v4..7), kb=0;
//             lanes 16-31 same with kb=8.
__device__ __forceinline__ v16bf load_fragA(const bf16_t* base, int lane, int ld) {
  int m  = lane & 15;
  int kb = (lane >> 4) << 3;            // 0 or 8
  const bf16_t* p = base + m * ld + kb;
  v8bf lo = *(const v8bf*)(p);          // K = kb .. kb+7
  v8bf hi = *(const v8bf*)(p + 16);     // K = kb+16 .. kb+23
  return __builtin_shufflevector(lo, hi, 0,1,2,3,4,5,6,7,8,9,10,11,12,13,14,15);
}

// B-matrix fragment (32x16 bf16) from transposed source B_t[n][k], ld elems.
// ISA layout: lanes 0-15 -> K kb..kb+7 and kb+16..kb+23, kb=0; lanes 16-31 kb=8.
__device__ __forceinline__ v16bf load_fragB(const bf16_t* baseT, int lane, int ld) {
  int n  = lane & 15;
  int kb = (lane >> 4) << 3;            // 0 or 8
  const bf16_t* p = baseT + n * ld + kb;
  v8bf lo = *(const v8bf*)(p);
  v8bf hi = *(const v8bf*)(p + 16);
  return __builtin_shufflevector(lo, hi, 0,1,2,3,4,5,6,7,8,9,10,11,12,13,14,15);
}

__device__ __forceinline__ v8f wmma_bf(v16bf a, v16bf b, v8f c) {
  return __builtin_amdgcn_wmma_f32_16x16x32_bf16(false, a, false, b, (short)0, c,
                                                 false, false);
}

// sum across the 8 lanes of an aligned 8-lane group (wave32)
__device__ __forceinline__ float grp8sum(float v) {
  v += __shfl_xor(v, 1);
  v += __shfl_xor(v, 2);
  v += __shfl_xor(v, 4);
  return v;
}

// ------------------------------- converts ----------------------------------
__global__ __launch_bounds__(256) void cvt_f32_bf16(const float* __restrict__ x,
                                                    bf16_t* __restrict__ y,
                                                    size_t n) {
  size_t i = (size_t)blockIdx.x * blockDim.x + threadIdx.x;
  size_t stride = (size_t)gridDim.x * blockDim.x;
  for (; i < n; i += stride) y[i] = f2bf(x[i]);
}

// --------------------------------- GEMM ------------------------------------
// C(f32, MxN) = A(bf16, MxK, row-major) @ B(bf16, KxN, row-major)
// block tile 64(M) x 128(N), K-step 32; 8 waves: wave -> (mt = w>>1, 4 n-tiles)
// Software pipelined: next K-step's global tiles staged in registers while the
// current K-step's WMMAs run; all 5 fragments loaded before the 4 WMMAs issue.
// out_mode 0: C row-major [m][n]
// out_mode 1: m = b*L+l, n = h*HD+hd -> C[((b*NH+h)*L+l)*HD+hd]
__global__ __launch_bounds__(256) void gemm_bf16(const bf16_t* __restrict__ A,
                                                 const bf16_t* __restrict__ B,
                                                 float* __restrict__ C,
                                                 int M, int N, int K,
                                                 int out_mode) {
  __shared__ bf16_t As[64 * 32];
  __shared__ bf16_t Bt[128 * 32];   // transposed tile [n][k]
  int tid = threadIdx.x, lane = tid & 31, wave = tid >> 5;
  int m0 = blockIdx.y * 64, n0 = blockIdx.x * 128;
  int mt = wave >> 1, ntb = (wave & 1) * 4;
  v8f acc[4] = {};

  int ar = tid >> 2, ak = (tid & 3) * 8;       // A: 64 rows x 32, 16B/thread
  int bk = tid >> 3, bn = (tid & 7) * 16;      // B: 32 rows x 128, 32B/thread
  const bf16_t* Aptr = A + (size_t)(m0 + ar) * K + ak;
  const bf16_t* Bptr = B + (size_t)bk * N + n0 + bn;

  // prologue: stage K-step 0
  v8bf aReg  = *(const v8bf*)(Aptr);
  v8bf bReg0 = *(const v8bf*)(Bptr);
  v8bf bReg1 = *(const v8bf*)(Bptr + 8);

  for (int k0 = 0; k0 < K; k0 += 32) {
    // commit staged registers to LDS
    *(v8bf*)&As[ar * 32 + ak] = aReg;
#pragma unroll
    for (int i = 0; i < 8; ++i) {
      Bt[(bn + i) * 32 + bk]     = bReg0[i];
      Bt[(bn + 8 + i) * 32 + bk] = bReg1[i];
    }
    __syncthreads();

    // stage next K-step from global while WMMAs run below
    if (k0 + 32 < K) {
      aReg  = *(const v8bf*)(Aptr + k0 + 32);
      bReg0 = *(const v8bf*)(Bptr + (size_t)(k0 + 32) * N);
      bReg1 = *(const v8bf*)(Bptr + (size_t)(k0 + 32) * N + 8);
    }
    if (k0 + 64 < K) {   // speculative L2 prefetch one more step ahead
      __builtin_prefetch(Aptr + k0 + 64, 0, 3);
      __builtin_prefetch(Bptr + (size_t)(k0 + 64) * N, 0, 3);
    }

    // load all fragments, then issue 4 back-to-back WMMAs
    v16bf af  = load_fragA(As + mt * 16 * 32, lane, 32);
    v16bf bf0 = load_fragB(Bt + (ntb + 0) * 16 * 32, lane, 32);
    v16bf bf1 = load_fragB(Bt + (ntb + 1) * 16 * 32, lane, 32);
    v16bf bf2 = load_fragB(Bt + (ntb + 2) * 16 * 32, lane, 32);
    v16bf bf3 = load_fragB(Bt + (ntb + 3) * 16 * 32, lane, 32);
    acc[0] = wmma_bf(af, bf0, acc[0]);
    acc[1] = wmma_bf(af, bf1, acc[1]);
    acc[2] = wmma_bf(af, bf2, acc[2]);
    acc[3] = wmma_bf(af, bf3, acc[3]);
    __syncthreads();
  }

  int colt = lane & 15, rowoff = (lane >> 4) << 3;
#pragma unroll
  for (int j = 0; j < 4; ++j) {
    int n = n0 + (ntb + j) * 16 + colt;
#pragma unroll
    for (int i = 0; i < 8; ++i) {
      int m = m0 + mt * 16 + rowoff + i;
      float v = acc[j][i];
      if (out_mode == 0) {
        C[(size_t)m * N + n] = v;
      } else {
        int b = m >> 11, l = m & (TL - 1);
        int h = n >> 6,  hd = n & (THD - 1);
        C[(((size_t)b * TNH + h) * TL + l) * THD + hd] = v;
      }
    }
  }
}

// --------------------------------- RoPE ------------------------------------
// net effect of permute_qk -> rope(half-rotate) -> undo_permute:
//   (x[2j], x[2j+1]) rotated by angle l * base^(-j/32)
__global__ __launch_bounds__(256) void rope_kernel(float* __restrict__ Q,
                                                   float* __restrict__ K) {
  size_t i = (size_t)blockIdx.x * blockDim.x + threadIdx.x;  // B*NH*L*32 threads
  int j = (int)(i & 31);
  size_t row = i >> 5;                       // (b*NH+h)*L + l
  int l = (int)(row & (TL - 1));
  float inv = __expf(-(float)j * 0.28782313662425572f);      // ln(10000)/32
  float ang = (float)l * inv;
  float s = __sinf(ang), c = __cosf(ang);
  size_t base = row * THD + 2 * j;
  float q0 = Q[base], q1 = Q[base + 1];
  Q[base]     = q0 * c - q1 * s;
  Q[base + 1] = q1 * c + q0 * s;
  float k0 = K[base], k1 = K[base + 1];
  K[base]     = k0 * c - k1 * s;
  K[base + 1] = k1 * c + k0 * s;
}

// --------------------------------- scan ------------------------------------
// one block (4 waves) per (b,h); sequential over 128 chunks of 16 tokens
__global__ __launch_bounds__(128) void ttt_scan(const float* __restrict__ XQ,
                                                const float* __restrict__ XK,
                                                const float* __restrict__ XV,
                                                const float* __restrict__ W1g,
                                                const float* __restrict__ b1g,
                                                const float* __restrict__ gw,
                                                const float* __restrict__ gbias,
                                                bf16_t* __restrict__ att) {
  __shared__ float  W1[64 * 64];        // state, f32
  __shared__ bf16_t W1bT[64 * 64];      // state transposed [g][f], bf16
  __shared__ float  b1s[64];
  __shared__ float  gvec[64], bvec[64];
  __shared__ float  xq[16 * 64], xk[16 * 64], xv[16 * 64];
  __shared__ bf16_t xqb[16 * 64], xkb[16 * 64];
  __shared__ float  Z1[16 * 64];
  __shared__ float  gradf[16 * 64];
  __shared__ bf16_t gradT[64 * 32];     // [g][k], k 16..31 stay zero (K pad)
  __shared__ float  Z1b[16 * 64];
  __shared__ float  attnS[16 * 16];
  __shared__ bf16_t Mb[16 * 32];        // [m][l], l 16..31 stay zero (K pad)

  int tid = threadIdx.x, lane = tid & 31, wave = tid >> 5;
  int bh = blockIdx.x;                  // b*NH + h
  int h = bh & (TNH - 1);
  int b = bh >> 5;
  const float last_eta = 1.0f / 16.0f;
  int colt = lane & 15, rowoff = (lane >> 4) << 3;

  for (int i = tid; i < 64 * 64; i += 128) {
    float w = W1g[(size_t)h * 64 * 64 + i];
    W1[i] = w;
    int f = i >> 6, g = i & 63;
    W1bT[g * 64 + f] = f2bf(w);
  }
  for (int i = tid; i < 64; i += 128) {
    b1s[i]  = b1g[h * 64 + i];
    gvec[i] = gw[h * 64 + i];
    bvec[i] = gbias[h * 64 + i];
  }
  for (int i = tid; i < 64 * 32; i += 128) gradT[i] = f2bf(0.0f);
  for (int i = tid; i < 16 * 32; i += 128) Mb[i] = f2bf(0.0f);
  __syncthreads();

  size_t hb = (size_t)bh * TL * THD;

  for (int t = 0; t < TL / TCS; ++t) {
    size_t off = hb + (size_t)t * TCS * THD;
    {   // stage chunk (16x64 each), 8 contiguous elems / thread
      int i0 = tid * 8;
#pragma unroll
      for (int i = 0; i < 8; ++i) {
        float q = XQ[off + i0 + i];
        float k = XK[off + i0 + i];
        float v = XV[off + i0 + i];
        xq[i0 + i] = q; xk[i0 + i] = k; xv[i0 + i] = v;
        xqb[i0 + i] = f2bf(q); xkb[i0 + i] = f2bf(k);
      }
      if (t + 1 < TL / TCS) {   // prefetch next chunk into L2/L0
        __builtin_prefetch(XQ + off + 1024 + i0, 0, 3);
        __builtin_prefetch(XK + off + 1024 + i0, 0, 3);
        __builtin_prefetch(XV + off + 1024 + i0, 0, 3);
      }
    }
    __syncthreads();

    // Z1 = xk @ W1 + b1 ; wave = n-tile
    {
      v16bf a0 = load_fragA(xkb, lane, 64);
      v16bf w0 = load_fragB(W1bT + wave * 16 * 64, lane, 64);
      v16bf a1 = load_fragA(xkb + 32, lane, 64);
      v16bf w1 = load_fragB(W1bT + wave * 16 * 64 + 32, lane, 64);
      v8f z = {};
      z = wmma_bf(a0, w0, z);
      z = wmma_bf(a1, w1, z);
      int col = wave * 16 + colt;
#pragma unroll
      for (int i = 0; i < 8; ++i)
        Z1[(rowoff + i) * 64 + col] = z[i] + b1s[col];
    }
    __syncthreads();

    // grad = ln_fused_l2_bwd(Z1, xv - xk, g, b); 8 lanes per row
    {
      int r = tid >> 3, c0 = (tid & 7) * 8;
      float x[8], tg[8];
      float s = 0.f;
#pragma unroll
      for (int i = 0; i < 8; ++i) {
        x[i]  = Z1[r * 64 + c0 + i];
        tg[i] = xv[r * 64 + c0 + i] - xk[r * 64 + c0 + i];
        s += x[i];
      }
      float mu = grp8sum(s) * (1.0f / 64.0f);
      float vs = 0.f;
#pragma unroll
      for (int i = 0; i < 8; ++i) { float d = x[i] - mu; vs += d * d; }
      float var = grp8sum(vs) * (1.0f / 64.0f);
      float std = __fsqrt_rn(var + 1e-6f);
      float rstd = 1.0f / std;
      float xh[8], gxh[8];
      float s1 = 0.f, s2 = 0.f;
#pragma unroll
      for (int i = 0; i < 8; ++i) {
        xh[i] = (x[i] - mu) * rstd;
        float go = gvec[c0 + i] * xh[i] + bvec[c0 + i] - tg[i];
        gxh[i] = go * gvec[c0 + i];
        s1 += gxh[i];
        s2 += gxh[i] * xh[i];
      }
      s1 = grp8sum(s1);
      s2 = grp8sum(s2);
      float sc = rstd * (1.0f / 64.0f);
#pragma unroll
      for (int i = 0; i < 8; ++i) {
        float gr = (64.0f * gxh[i] - s1 - xh[i] * s2) * sc;
        gradf[r * 64 + c0 + i] = gr;
        gradT[(c0 + i) * 32 + r] = f2bf(gr);
      }
    }
    __syncthreads();

    // attn = xq @ xk^T (16x16); B_t for xk^T is xk row-major
    if (wave == 0) {
      v16bf qa0 = load_fragA(xqb, lane, 64);
      v16bf kb0 = load_fragB(xkb, lane, 64);
      v16bf qa1 = load_fragA(xqb + 32, lane, 64);
      v16bf kb1 = load_fragB(xkb + 32, lane, 64);
      v8f a = {};
      a = wmma_bf(qa0, kb0, a);
      a = wmma_bf(qa1, kb1, a);
#pragma unroll
      for (int i = 0; i < 8; ++i)
        attnS[(rowoff + i) * 16 + colt] = a[i];
    }
    __syncthreads();

    // M[m][l] = (l<=m) * eta[m]*(attn+1), folds tril_eta@grad and eta*attn@grad
    for (int idx = tid; idx < 256; idx += 128) {
      int m = idx >> 4, l2 = idx & 15;
      float v = (l2 <= m) ? (attnS[m * 16 + l2] + 1.0f) / (float)(m + 1) : 0.0f;
      Mb[m * 32 + l2] = f2bf(v);
    }
    __syncthreads();

    // Z1_bar = xq @ W1 - M @ grad + b1
    {
      v16bf qa0 = load_fragA(xqb, lane, 64);
      v16bf w0  = load_fragB(W1bT + wave * 16 * 64, lane, 64);
      v16bf qa1 = load_fragA(xqb + 32, lane, 64);
      v16bf w1  = load_fragB(W1bT + wave * 16 * 64 + 32, lane, 64);
      v16bf ma  = load_fragA(Mb, lane, 32);
      v16bf gb  = load_fragB(gradT + wave * 16 * 32, lane, 32);
      v8f zb = {};
      zb = wmma_bf(qa0, w0, zb);
      zb = wmma_bf(qa1, w1, zb);
      v8f mg = {};
      mg = wmma_bf(ma, gb, mg);
      int col = wave * 16 + colt;
#pragma unroll
      for (int i = 0; i < 8; ++i)
        Z1b[(rowoff + i) * 64 + col] = zb[i] - mg[i] + b1s[col];
    }
    __syncthreads();

    // out = xq + ln_fwd(Z1_bar); store bf16 at (b, l, h*64+c)
    {
      int r = tid >> 3, c0 = (tid & 7) * 8;
      float x[8];
      float s = 0.f;
#pragma unroll
      for (int i = 0; i < 8; ++i) { x[i] = Z1b[r * 64 + c0 + i]; s += x[i]; }
      float mu = grp8sum(s) * (1.0f / 64.0f);
      float vs = 0.f;
#pragma unroll
      for (int i = 0; i < 8; ++i) { float d = x[i] - mu; vs += d * d; }
      float var = grp8sum(vs) * (1.0f / 64.0f);
      float rstd = __frsqrt_rn(var + 1e-6f);
      v8bf o;
#pragma unroll
      for (int i = 0; i < 8; ++i) {
        float xh = (x[i] - mu) * rstd;
        o[i] = f2bf(gvec[c0 + i] * xh + bvec[c0 + i] + xq[r * 64 + c0 + i]);
      }
      size_t l = (size_t)t * TCS + r;
      *(v8bf*)&att[((size_t)b * TL + l) * TD + h * 64 + c0] = o;
    }

    // W1 -= last_eta * xk^T @ grad ; wave = f-tile (rows 16w..16w+15)
    {
      int m = wave * 16 + colt;
      int kb = (lane >> 4) << 3;        // 0 or 8 (A-layout low K), high K = pad
      v16bf axt;
#pragma unroll
      for (int i = 0; i < 8; ++i) {
        axt[i]     = xkb[(kb + i) * 64 + m];   // (f=m, k) = xk[k][m], k<16
        axt[8 + i] = f2bf(0.0f);               // K 16..31 padded with zero
      }
      v16bf g0 = load_fragB(gradT + 0 * 16 * 32, lane, 32);
      v16bf g1 = load_fragB(gradT + 1 * 16 * 32, lane, 32);
      v16bf g2 = load_fragB(gradT + 2 * 16 * 32, lane, 32);
      v16bf g3 = load_fragB(gradT + 3 * 16 * 32, lane, 32);
      v8f p0 = {}, p1 = {}, p2 = {}, p3 = {};
      p0 = wmma_bf(axt, g0, p0);
      p1 = wmma_bf(axt, g1, p1);
      p2 = wmma_bf(axt, g2, p2);
      p3 = wmma_bf(axt, g3, p3);
      v8f pj[4] = {p0, p1, p2, p3};
#pragma unroll
      for (int j = 0; j < 4; ++j) {
#pragma unroll
        for (int i = 0; i < 8; ++i) {
          int f = wave * 16 + rowoff + i;
          int g = j * 16 + colt;
          float w = W1[f * 64 + g] - last_eta * pj[j][i];
          W1[f * 64 + g] = w;
          W1bT[g * 64 + f] = f2bf(w);
        }
      }
    }
    // b1 -= last_eta * sum_k grad[k][:]
    if (tid < 64) {
      float s = 0.f;
#pragma unroll
      for (int r = 0; r < 16; ++r) s += gradf[r * 64 + tid];
      b1s[tid] -= last_eta * s;
    }
    __syncthreads();
  }
}

// ------------------------------- launcher ----------------------------------
extern "C" void kernel_launch(void* const* d_in, const int* in_sizes, int n_in,
                              void* d_out, int out_size, void* d_ws,
                              size_t ws_size, hipStream_t stream) {
  (void)in_sizes; (void)n_in; (void)out_size; (void)ws_size;
  const float* hs = (const float*)d_in[0];
  const float* wq = (const float*)d_in[1];
  const float* wk = (const float*)d_in[2];
  const float* wv = (const float*)d_in[3];
  const float* wo = (const float*)d_in[4];
  const float* W1 = (const float*)d_in[5];
  const float* b1 = (const float*)d_in[6];
  const float* gw = (const float*)d_in[7];
  const float* gb = (const float*)d_in[8];

  const size_t M = (size_t)TB * TL;          // 16384
  const size_t DD = (size_t)TD;              // 2048

  char* ws = (char*)d_ws;
  size_t off = 0;
  auto carve = [&](size_t bytes) -> void* {
    void* p = ws + off;
    off += (bytes + 255) & ~(size_t)255;
    return p;
  };
  bf16_t* hsb  = (bf16_t*)carve(M * DD * sizeof(bf16_t));
  bf16_t* wqb  = (bf16_t*)carve(DD * DD * sizeof(bf16_t));
  bf16_t* wkb  = (bf16_t*)carve(DD * DD * sizeof(bf16_t));
  bf16_t* wvb  = (bf16_t*)carve(DD * DD * sizeof(bf16_t));
  bf16_t* wob  = (bf16_t*)carve(DD * DD * sizeof(bf16_t));
  float*  XQ   = (float*)carve(M * DD * sizeof(float));
  float*  XK   = (float*)carve(M * DD * sizeof(float));
  float*  XV   = (float*)carve(M * DD * sizeof(float));
  bf16_t* attb = (bf16_t*)carve(M * DD * sizeof(bf16_t));

  cvt_f32_bf16<<<4096, 256, 0, stream>>>(hs, hsb, M * DD);
  cvt_f32_bf16<<<1024, 256, 0, stream>>>(wq, wqb, DD * DD);
  cvt_f32_bf16<<<1024, 256, 0, stream>>>(wk, wkb, DD * DD);
  cvt_f32_bf16<<<1024, 256, 0, stream>>>(wv, wvb, DD * DD);
  cvt_f32_bf16<<<1024, 256, 0, stream>>>(wo, wob, DD * DD);

  dim3 gg(TD / 128, (unsigned)(M / 64));
  gemm_bf16<<<gg, 256, 0, stream>>>(hsb, wqb, XQ, (int)M, TD, TD, 1);
  gemm_bf16<<<gg, 256, 0, stream>>>(hsb, wkb, XK, (int)M, TD, TD, 1);
  gemm_bf16<<<gg, 256, 0, stream>>>(hsb, wvb, XV, (int)M, TD, TD, 1);

  rope_kernel<<<(unsigned)((size_t)TB * TNH * TL * (THD / 2) / 256), 256, 0,
                stream>>>(XQ, XK);

  ttt_scan<<<TB * TNH, 128, 0, stream>>>(XQ, XK, XV, W1, b1, gw, gb, attb);

  gemm_bf16<<<gg, 256, 0, stream>>>(attb, wob, (float*)d_out, (int)M, TD, TD, 0);
}
